// RNN_78091095376442
// MI455X (gfx1250) — matrix operand order repaired and checked by reading, not compiled, not run on hardware
//
#include <hip/hip_runtime.h>
#include <hip/hip_bf16.h>

// ---------------------------------------------------------------------------
// Problem constants (reference: T=512, B=64, I=H=1024, L=2)
// ---------------------------------------------------------------------------
#define TT 512
#define BB 64
#define II 1024
#define HH 1024

// Bulk-GEMM operand staging: 2 = TDM tensor_load_to_lds (CDNA5 DMA engine),
//                            1 = per-lane async global->LDS b128, 0 = sync LDS stores
#define STAGE_MODE 2

typedef __attribute__((ext_vector_type(16))) __bf16        v16bf;
typedef __attribute__((ext_vector_type(8)))  float         v8f;
typedef __attribute__((ext_vector_type(8)))  unsigned short us8;
typedef __attribute__((ext_vector_type(4)))  unsigned int   u32x4;
typedef __attribute__((ext_vector_type(8)))  unsigned int   u32x8;

#define WMMA_BF16(a, b, c) \
    __builtin_amdgcn_wmma_f32_16x16x32_bf16(false, (a), false, (b), (short)0, (c), false, false)

// A-fragment (16x32 bf16, MxK): per lane two contiguous 16-byte chunks.
__device__ __forceinline__ v16bf load_frag2(const __bf16* p0, const __bf16* p1) {
    union { us8 h[2]; v16bf v; } u;
    u.h[0] = *(const us8*)p0;
    u.h[1] = *(const us8*)p1;
    return u.v;
}
// B-fragment (32x16 bf16, KxN): per lane one contiguous 32-byte run.
__device__ __forceinline__ v16bf load_frag16(const __bf16* p) {
    return load_frag2(p, p + 8);
}

// Per-lane async global->LDS copy (fallback path). Tracked by ASYNCcnt.
__device__ __forceinline__ void async_cp16(unsigned lds_off, const __bf16* gsrc) {
    asm volatile("global_load_async_to_lds_b128 %0, %1, off"
                 :: "v"(lds_off), "v"((unsigned long long)(size_t)gsrc)
                 : "memory");
}
#define WAIT_ASYNC(n) asm volatile("s_wait_asynccnt " #n ::: "memory")

// TDM 2D tile load: rows x 32-col bf16 tile, tensor row stride 1024 elems,
// LDS destination padded 4 DWORDs per 16 DWORDs (-> 80 B rows, bank-safe).
// D# per CDNA5 ISA 8.3/8.4; 2D => groups 2/3 omitted. Wave-level DMA:
// caller must ensure only ONE wave executes this (TDM ignores EXEC).
__device__ __forceinline__ void tdm_load_2d(unsigned lds_addr, const void* gsrc,
                                            unsigned tile_rows) {
    const unsigned long long ga = (unsigned long long)(size_t)gsrc;
    u32x4 g0;
    g0[0] = 1u;                                            // count=1, user descriptor
    g0[1] = lds_addr;                                      // lds_addr [63:32]
    g0[2] = (unsigned)ga;                                  // global_addr [95:64]
    g0[3] = (unsigned)((ga >> 32) & 0x01ffffffu)           // global_addr [120:96]
          | (2u << 30);                                    // type=2 ("image")
    u32x8 g1;
    g1[0] = (1u << 16)                                     // data_size=1 (2 bytes)
          | (1u << 20)                                     // pad_enable
          | (3u << 22)                                     // pad_interval: 16 DWORDs
          | (3u << 25);                                    // pad_amount: 4 DWORDs
    g1[1] = (1024u & 0xffffu) << 16;                       // tensor_dim0 lo16 (=1024)
    g1[2] = (1024u >> 16) | ((tile_rows & 0xffffu) << 16); // dim0 hi16 | tensor_dim1 lo16
    g1[3] = (tile_rows >> 16) | (32u << 16);               // dim1 hi16 | tile_dim0=32
    g1[4] = tile_rows & 0xffffu;                           // tile_dim1 | tile_dim2=0
    g1[5] = 1024u;                                         // tensor_dim0_stride lo32
    g1[6] = 0u;                                            // stride0 hi16 | stride1 lo16
    g1[7] = 0u;                                            // stride1 hi32
    asm volatile("tensor_load_to_lds %0, %1" :: "s"(g0), "s"(g1) : "memory");
}

// ---------------------------------------------------------------------------
// Kernel 0: fp32 -> bf16 conversion (x and w_l0), vectorized x4
// ---------------------------------------------------------------------------
__global__ __launch_bounds__(256) void rnn_cvt_kernel(const float* __restrict__ src,
                                                      __bf16* __restrict__ dst, int n4) {
    const int i = blockIdx.x * blockDim.x + threadIdx.x;
    if (i < n4) {
        const float4 f = ((const float4*)src)[i];
        ushort4 u;
        u.x = __builtin_bit_cast(unsigned short, (__bf16)f.x);
        u.y = __builtin_bit_cast(unsigned short, (__bf16)f.y);
        u.z = __builtin_bit_cast(unsigned short, (__bf16)f.z);
        u.w = __builtin_bit_cast(unsigned short, (__bf16)f.w);
        ((ushort4*)dst)[i] = u;
    }
}

// ---------------------------------------------------------------------------
// Kernel 1: init — seed h-state ping-pong buffers from `state`, reset barrier
// ---------------------------------------------------------------------------
__global__ void rnn_init_kernel(__bf16* __restrict__ h0buf, __bf16* __restrict__ h1buf,
                                unsigned* __restrict__ sync, const float* __restrict__ state) {
    int i = blockIdx.x * blockDim.x + threadIdx.x;
    const int n = BB * HH;
    if (i < n) {
        h0buf[i]     = (__bf16)0.0f;
        h1buf[i]     = (__bf16)0.0f;
        h0buf[n + i] = (__bf16)state[i];        // buffer 1 holds h_{-1} = state[layer]
        h1buf[n + i] = (__bf16)state[n + i];
    }
    if (i < 2) sync[i] = 0u;
}

// ---------------------------------------------------------------------------
// Kernel 2: bulk input projection  xin0 = x @ w_l0^T   (32768 x 1024 x 1024)
// 128 threads = 4 waves; WG tile 128x64; wave tile 32x64 (8 accumulators).
// Operands staged bf16 into double-buffered LDS via TDM (or async copies).
// LDS row stride 40 elems (80 B): 16-B chunk aligned and bank-conflict free.
// ---------------------------------------------------------------------------
#define KP 40
#define X_ELEMS (128 * KP)            // x tile elements per buffer
#define BUF_ELEMS ((128 + 64) * KP)   // x tile + w tile per buffer
__global__ __launch_bounds__(128) void rnn_proj_kernel(const __bf16* __restrict__ xbf,
                                                       const __bf16* __restrict__ wbf,
                                                       float* __restrict__ xin) {
    extern __shared__ __bf16 lds[];   // 2 buffers x (128+64) x KP bf16 = 30720 B
    const int m0 = (blockIdx.x >> 4) * 128;
    const int n0 = (blockIdx.x & 15) * 64;
    const int tid = threadIdx.x;
    const int wv = tid >> 5;
    const int lane = tid & 31;
    const int l16 = lane & 15;
    const bool lo = lane < 16;

    const __bf16* xg0 = xbf + (long)m0 * II;
    const __bf16* wg0 = wbf + (long)n0 * II;
    const unsigned ldsbase = (unsigned)(size_t)(void*)lds;

#if STAGE_MODE == 2
    // One TDM descriptor per operand tile; issued by wave 0 only.
    auto stage = [&](int kc, int b) {
        const unsigned xo = ldsbase + (unsigned)(b * BUF_ELEMS) * 2u;
        const unsigned wo = xo + (unsigned)X_ELEMS * 2u;
        const long kb = (long)kc * 32;
        tdm_load_2d(xo, xg0 + kb, 128u);
        tdm_load_2d(wo, wg0 + kb, 64u);
    };
#elif STAGE_MODE == 1
    auto stage = [&](int kc, int b) {
        const unsigned xo = ldsbase + (unsigned)(b * BUF_ELEMS) * 2u;
        const unsigned wo = xo + (unsigned)X_ELEMS * 2u;
        const long kb = (long)kc * 32;
        #pragma unroll
        for (int j = 0; j < 4; ++j) {                   // 128 rows x 4 16-B chunks
            const int i = tid + j * 128;
            const int r = i >> 2, cc = (i & 3) * 8;
            async_cp16(xo + (unsigned)(r * KP + cc) * 2u, xg0 + r * (long)II + kb + cc);
        }
        #pragma unroll
        for (int j = 0; j < 2; ++j) {                   // 64 rows x 4 16-B chunks
            const int i = tid + j * 128;
            const int r = i >> 2, cc = (i & 3) * 8;
            async_cp16(wo + (unsigned)(r * KP + cc) * 2u, wg0 + r * (long)II + kb + cc);
        }
    };
#else
    auto stage = [&](int kc, int b) {
        __bf16* xo = lds + b * BUF_ELEMS;
        __bf16* wo = xo + X_ELEMS;
        const long kb = (long)kc * 32;
        for (int i = tid; i < 128 * 32; i += 128) {
            const int r = i >> 5, c = i & 31;
            xo[r * KP + c] = xg0[r * (long)II + kb + c];
        }
        for (int i = tid; i < 64 * 32; i += 128) {
            const int r = i >> 5, c = i & 31;
            wo[r * KP + c] = wg0[r * (long)II + kb + c];
        }
    };
#endif

    v8f acc[2][4] = {};
#if STAGE_MODE == 2
    if (tid < 32) stage(0, 0);                          // wave 0 issues the DMA
#else
    stage(0, 0);
#endif
    #pragma unroll 1                                    // keep one register copy of the body
    for (int kc = 0; kc < 32; ++kc) {
        const int cur = kc & 1;
#if STAGE_MODE == 2
        if (tid < 32) {                                 // s_cbranch_execz skip for waves 1..3
            if (kc + 1 < 32) {
                stage(kc + 1, cur ^ 1);
                __builtin_amdgcn_s_wait_tensorcnt(2);   // retire chunk kc's 2 DMAs
            } else {
                __builtin_amdgcn_s_wait_tensorcnt(0);
            }
        }
#elif STAGE_MODE == 1
        if (kc + 1 < 32) { stage(kc + 1, cur ^ 1); WAIT_ASYNC(6); }
        else             { WAIT_ASYNC(0); }
#else
        if (kc + 1 < 32) stage(kc + 1, cur ^ 1);
#endif
        __syncthreads();                                // chunk kc visible to all waves
        const __bf16* xb = lds + cur * BUF_ELEMS;
        const __bf16* wb = xb + X_ELEMS;
        #pragma unroll
        for (int p = 0; p < 2; ++p) {                   // two 16-row M-subtiles per wave
            const __bf16* ap = xb + (wv * 32 + p * 16 + l16) * KP + (lo ? 0 : 8);
            const v16bf a = load_frag2(ap, ap + 16);
            #pragma unroll
            for (int t = 0; t < 4; ++t) {
                const v16bf b = load_frag2(wb + (t * 16 + l16) * KP + (lo ? 0 : 8), wb + (t * 16 + l16) * KP + (lo ? 16 : 24));
                acc[p][t] = WMMA_BF16(a, b, acc[p][t]);
            }
        }
        __syncthreads();                                // done reading before refill
    }
    #pragma unroll
    for (int p = 0; p < 2; ++p)
        #pragma unroll
        for (int t = 0; t < 4; ++t)
            #pragma unroll
            for (int v = 0; v < 8; ++v) {
                const int m = m0 + wv * 32 + p * 16 + v + (lo ? 0 : 8);
                const int n = n0 + t * 16 + l16;
                xin[(long)m * HH + n] = acc[p][t][v];
            }
}

// ---------------------------------------------------------------------------
// Grid-wide barrier (persistent kernel; 96 WGs guaranteed co-resident).
// ---------------------------------------------------------------------------
__device__ __forceinline__ void grid_sync(unsigned* cnt, unsigned* gen,
                                          unsigned nwg, unsigned s) {
    __syncthreads();
    if (threadIdx.x == 0) {
        const unsigned prev =
            __hip_atomic_fetch_add(cnt, 1u, __ATOMIC_ACQ_REL, __HIP_MEMORY_SCOPE_AGENT);
        if (prev == nwg - 1u) {
            __hip_atomic_store(cnt, 0u, __ATOMIC_RELAXED, __HIP_MEMORY_SCOPE_AGENT);
            __hip_atomic_fetch_add(gen, 1u, __ATOMIC_ACQ_REL, __HIP_MEMORY_SCOPE_AGENT);
        } else {
            while (__hip_atomic_load(gen, __ATOMIC_ACQUIRE, __HIP_MEMORY_SCOPE_AGENT) <= s)
                __builtin_amdgcn_s_sleep(1);
        }
    }
    __syncthreads();
}

// ---------------------------------------------------------------------------
// Kernel 3: persistent two-layer wavefront scan.
//   WG 0..31  (layer 0): superstep s computes h0_s   = tanh(xin0_s + h0_{s-1} Ws0^T + b0)
//   WG 32..95 (layer 1): superstep s computes h1_{s-1}= tanh(h0_{s-1} Wl1^T + h1_{s-2} Ws1^T + b1)
// Weight strips live in LDS (bf16, stride 1032 elems) for the whole scan.
// h buffers ping-pong in global scratch (L2-resident, 128 KB each).
// ---------------------------------------------------------------------------
#define NWG_SCAN 96u
#define WS 1032   // padded LDS row stride (elements)
__global__ __launch_bounds__(128) void rnn_scan_kernel(
    const float* __restrict__ ws0, const float* __restrict__ bs0,
    const float* __restrict__ wl1, const float* __restrict__ ws1,
    const float* __restrict__ bs1, const float* __restrict__ xin,
    __bf16* __restrict__ h0buf, __bf16* __restrict__ h1buf,
    float* __restrict__ out, unsigned* __restrict__ cnt, unsigned* __restrict__ gen) {
    extern __shared__ __bf16 wlds[];   // 32 x WS bf16 = 66048 B (< 320 KB/WGP)
    const int wg = blockIdx.x;
    const int tid = threadIdx.x;
    const int wv = tid >> 5;
    const int lane = tid & 31;
    const int l16 = lane & 15;
    const bool lo = lane < 16;
    const bool is_l0 = wg < 32;
    const int HB = BB * HH;

    // One-time weight strip preload, fp32 -> bf16.
    if (is_l0) {
        const int n0 = wg * 32;
        for (int i = tid; i < 32 * HH; i += 128) {
            const int r = i >> 10, c = i & 1023;
            wlds[r * WS + c] = (__bf16)ws0[(long)(n0 + r) * HH + c];
        }
    } else {
        const int n0 = (wg - 32) * 16;
        for (int i = tid; i < 16 * HH; i += 128) {
            const int r = i >> 10, c = i & 1023;
            wlds[r * WS + c]        = (__bf16)wl1[(long)(n0 + r) * HH + c];
            wlds[(16 + r) * WS + c] = (__bf16)ws1[(long)(n0 + r) * HH + c];
        }
    }
    __syncthreads();

    const int mrowA = wv * 16 + l16;   // A-operand row for this lane (batch row 0..63)
    for (int s = 0; s <= TT; ++s) {
        if (is_l0 && s < TT) {
            const int n0 = wg * 32;
            // Pull next timestep's xin tile toward L2 while this step computes.
            if (s + 1 < TT)
                __builtin_prefetch(&xin[((long)(s + 1) * BB + mrowA) * HH + n0], 0, 1);
            const __bf16* hprev = h0buf + ((s + 1) & 1) * HB;   // h0_{s-1}
            v8f acc[2];
            #pragma unroll
            for (int t = 0; t < 2; ++t) {
                const int n = n0 + t * 16 + l16;
                const float bias = bs0[n];
                #pragma unroll
                for (int v = 0; v < 8; ++v) {
                    const int m = wv * 16 + v + (lo ? 0 : 8);
                    acc[t][v] = xin[((long)s * BB + m) * HH + n] + bias;
                }
            }
            for (int kb = 0; kb < HH; kb += 32) {
                const __bf16* ap = hprev + mrowA * HH + kb + (lo ? 0 : 8);
                const v16bf a = load_frag2(ap, ap + 16);
                #pragma unroll
                for (int t = 0; t < 2; ++t) {
                    const v16bf b = load_frag2(wlds + (t * 16 + l16) * WS + kb + (lo ? 0 : 8), wlds + (t * 16 + l16) * WS + kb + (lo ? 16 : 24));
                    acc[t] = WMMA_BF16(a, b, acc[t]);
                }
            }
            __bf16* hdst = h0buf + (s & 1) * HB;                // h0_s
            #pragma unroll
            for (int t = 0; t < 2; ++t)
                #pragma unroll
                for (int v = 0; v < 8; ++v) {
                    const int m = wv * 16 + v + (lo ? 0 : 8);
                    const int n = n0 + t * 16 + l16;
                    const float y = tanhf(acc[t][v]);
                    hdst[m * HH + n] = (__bf16)y;
                    if (s == TT - 1)                            // hl0 (final state, layer 0)
                        out[(long)TT * HB + (long)m * HH + n] = y;
                }
        } else if (!is_l0 && s >= 1) {
            const int n0 = (wg - 32) * 16;
            const __bf16* h0src = h0buf + ((s + 1) & 1) * HB;   // h0_{s-1}
            const __bf16* h1src = h1buf + (s & 1) * HB;         // h1_{s-2}
            v8f acc;
            {
                const float bias = bs1[n0 + l16];
                #pragma unroll
                for (int v = 0; v < 8; ++v) acc[v] = bias;
            }
            for (int kb = 0; kb < HH; kb += 32) {               // input projection (K 0..1023)
                const __bf16* ap = h0src + mrowA * HH + kb + (lo ? 0 : 8);
                const v16bf a = load_frag2(ap, ap + 16);
                const v16bf b = load_frag2(wlds + l16 * WS + kb + (lo ? 0 : 8), wlds + l16 * WS + kb + (lo ? 16 : 24));
                acc = WMMA_BF16(a, b, acc);
            }
            for (int kb = 0; kb < HH; kb += 32) {               // recurrence (K 1024..2047)
                const __bf16* ap = h1src + mrowA * HH + kb + (lo ? 0 : 8);
                const v16bf a = load_frag2(ap, ap + 16);
                const v16bf b = load_frag2(wlds + (16 + l16) * WS + kb + (lo ? 0 : 8), wlds + (16 + l16) * WS + kb + (lo ? 16 : 24));
                acc = WMMA_BF16(a, b, acc);
            }
            __bf16* hdst = h1buf + ((s + 1) & 1) * HB;          // h1_{s-1}
            const long tout = s - 1;
            #pragma unroll
            for (int v = 0; v < 8; ++v) {
                const int m = wv * 16 + v + (lo ? 0 : 8);
                const int n = n0 + l16;
                const float y = tanhf(acc[v]);
                hdst[m * HH + n] = (__bf16)y;
                out[(tout * BB + m) * (long)HH + n] = y;        // hs1[t]
                if (s == TT)                                    // hl1 (final state, layer 1)
                    out[(long)(TT + 1) * HB + (long)m * HH + n] = y;
            }
        }
        if (s < TT) grid_sync(cnt, gen, NWG_SCAN, (unsigned)s);
    }
}

// ---------------------------------------------------------------------------
// Launcher
// ---------------------------------------------------------------------------
extern "C" void kernel_launch(void* const* d_in, const int* in_sizes, int n_in,
                              void* d_out, int out_size, void* d_ws, size_t ws_size,
                              hipStream_t stream) {
    const float* x     = (const float*)d_in[0];
    const float* state = (const float*)d_in[1];
    const float* wl0   = (const float*)d_in[2];
    const float* ws0   = (const float*)d_in[3];
    const float* bs0   = (const float*)d_in[4];
    const float* wl1   = (const float*)d_in[5];
    const float* ws1   = (const float*)d_in[6];
    const float* bs1   = (const float*)d_in[7];
    float* out = (float*)d_out;

    // Workspace layout:
    //   xin0 fp32 (128 MiB) | x bf16 (64 MiB) | w_l0 bf16 (2 MiB)
    //   | 2x h0 ping-pong bf16 | 2x h1 ping-pong bf16 | 2x u32 barrier
    const size_t XIN_BYTES = (size_t)TT * BB * HH * sizeof(float);
    float*    xin   = (float*)d_ws;
    __bf16*   xbf   = (__bf16*)((char*)d_ws + XIN_BYTES);
    __bf16*   wbf   = xbf + (size_t)TT * BB * II;
    __bf16*   h0buf = wbf + (size_t)HH * II;
    __bf16*   h1buf = h0buf + 2 * BB * HH;
    unsigned* syn   = (unsigned*)(h1buf + 2 * BB * HH);

    rnn_init_kernel<<<(BB * HH + 255) / 256, 256, 0, stream>>>(h0buf, h1buf, syn, state);

    const int xn4 = TT * BB * II / 4, wn4 = HH * II / 4;
    rnn_cvt_kernel<<<(xn4 + 255) / 256, 256, 0, stream>>>(x, xbf, xn4);
    rnn_cvt_kernel<<<(wn4 + 255) / 256, 256, 0, stream>>>(wl0, wbf, wn4);

    const size_t PROJ_LDS = 2u * BUF_ELEMS * sizeof(__bf16);  // 30720 B
    rnn_proj_kernel<<<(TT * BB / 128) * (HH / 64), 128, PROJ_LDS, stream>>>(xbf, wbf, xin);

    const size_t SCAN_LDS = 32u * WS * sizeof(__bf16);        // 66048 B
    rnn_scan_kernel<<<NWG_SCAN, 128, SCAN_LDS, stream>>>(ws0, bs0, wl1, ws1, bs1,
                                                         xin, h0buf, h1buf, out,
                                                         syn, syn + 1);
}
